// CTM_partpad_dict_BN_62646392979522
// MI455X (gfx1250) — compile-verified, matrix operand order checked by reading, not run on hardware
//
#include <hip/hip_runtime.h>
#include <hip/hip_bf16.h>

// ---------------- problem constants (from setup_inputs) ----------------
#define BB    32
#define N0    12288
#define CIN   64
#define COUT  128
#define HH    128
#define WW    96
#define HN    64
#define WN    48
#define NS    (HN*WN)          // 3072
#define EPS   1e-6f
#define BN_EPS 1e-5f
#define NPTS  (BB*N0)          // 393216
#define MROWS (BB*N0)          // GEMM rows for tok
#define MCONV (BB*HN*WN)       // 98304 conv output positions

#define CONV_MT 4              // m-tiles per conv block
#define SKIP_MT 8              // m-tiles per skip-GEMM block

// ---------------- WMMA types ----------------
typedef __attribute__((ext_vector_type(16))) __bf16 v16bf;
typedef __attribute__((ext_vector_type(8)))  float  v8f;

union Frag16 { v16bf v; unsigned short u[16]; };

__device__ inline unsigned short f2bf(float f) {
  unsigned int u = __float_as_uint(f);
  unsigned int r = u + 0x7FFFu + ((u >> 16) & 1u);   // round-to-nearest-even
  return (unsigned short)(r >> 16);
}

// LDS byte address of a __shared__ object (low 32 bits of the generic ptr)
__device__ inline unsigned lds_addr(const void* p) {
  return (unsigned)(size_t)p;
}

// async global->LDS copy of 16B per lane (gfx1250, ASYNCcnt-tracked)
__device__ inline void async_g2l_b128(unsigned lds_off, unsigned gl_off,
                                      const float* sbase) {
  asm volatile("global_load_async_to_lds_b128 %0, %1, %2 offset:0"
               :: "v"(lds_off), "v"(gl_off), "s"(sbase) : "memory");
}
__device__ inline void wait_async0() {
  asm volatile("s_wait_asynccnt 0x0" ::: "memory");
}
__device__ inline void wait_async4() {   // async loads retire in order:
  asm volatile("s_wait_asynccnt 0x4" ::: "memory");  // <=4 left => prev chunk done
}

// ---------------- index helpers (match reference exactly) ----------------
__device__ inline int grid_idx(float lx, float ly, int Hh, int Ww) {
  lx = fminf(fmaxf(lx, -1.f), 1.f);
  ly = fminf(fmaxf(ly, -1.f), 1.f);
  int px = (int)rintf(0.5f * (lx + 1.f) * (float)Ww - 0.5f);
  int py = (int)rintf(0.5f * (ly + 1.f) * (float)Hh - 0.5f);
  px = min(max(px, 0), Ww - 1);
  py = min(max(py, 0), Hh - 1);
  return py * Ww + px;
}

__device__ inline void bilin4(float lx, float ly, int Hh, int Ww,
                              int* i4, float* w4) {
  lx = fminf(fmaxf(lx, -1.f), 1.f);
  ly = fminf(fmaxf(ly, -1.f), 1.f);
  float xs = 0.5f * (lx + 1.f) * (float)Ww - 0.5f;
  float ys = 0.5f * (ly + 1.f) * (float)Hh - 0.5f;
  int xl = (int)floorf(xs); xl = min(max(xl, 0), Ww - 1);
  int xh = min(xl + 1, Ww - 1);
  int yl = (int)floorf(ys); yl = min(max(yl, 0), Hh - 1);
  int yh = min(yl + 1, Hh - 1);
  float xw = fminf((float)xh, xs) - (float)xl;
  float yw = fminf((float)yh, ys) - (float)yl;
  w4[0] = (1.f - xw) * (1.f - yw);
  w4[1] = xw * (1.f - yw);
  w4[2] = (1.f - xw) * yw;
  w4[3] = xw * yw;
  i4[0] = yl * Ww + xl; i4[1] = yl * Ww + xh;
  i4[2] = yh * Ww + xl; i4[3] = yh * Ww + xh;
}

// ================= kernel 1: pixel hit count =================
__global__ void k_count(const float* __restrict__ loc, float* __restrict__ cnt) {
  int t = blockIdx.x * blockDim.x + threadIdx.x;
  if (t >= NPTS) return;
  int b = t / N0;
  int pix = grid_idx(loc[2 * t], loc[2 * t + 1], HH, WW);
  atomicAdd(&cnt[b * (HH * WW) + pix], 1.0f);
}

// ================= kernel 2: token2map scatter (mean) =================
__global__ void k_scatter_fmap(const float* __restrict__ x,
                               const float* __restrict__ loc,
                               const int* __restrict__ idx_agg,
                               const float* __restrict__ cnt,
                               float* __restrict__ fmap) {
  int t = blockIdx.x;            // one orig point per block
  int c = threadIdx.x;           // channel 0..63
  int b = t / N0;
  int pix = grid_idx(loc[2 * t], loc[2 * t + 1], HH, WW);
  float w = 1.0f / (cnt[b * (HH * WW) + pix] + EPS);
  int ia = idx_agg[t];
  float v = x[((long)b * N0 + ia) * CIN + c] * w;
  atomicAdd(&fmap[((long)b * (HH * WW) + pix) * CIN + c], v);
}

// ================= kernel 3: 3x3 s2 conv as implicit GEMM (WMMA bf16) =====
// y[m, co] = sum_{k} im2col(fmap)[m, k] * cw[k, co] + cb,  K = 9*64 = 576.
// Weight K-chunk (32x128 f32 = 16KB, contiguous: k = kk*32+r) is staged in
// LDS with async global->LDS loads, double-buffered & pipelined on ASYNCcnt.
// Block = CONV_MT m-tiles x 8 n-tiles (one n-tile per wave); the B fragment
// is built once per K-chunk and reused across CONV_MT WMMAs.
__global__ void k_conv_wmma(const float* __restrict__ fmap,   // [B,128,96,64]
                            const float* __restrict__ cw,     // [3,3,64,128]
                            const float* __restrict__ cb,     // [128]
                            float* __restrict__ y) {          // [MCONV,128]
  __shared__ float wlds[2][32 * COUT];              // 2 x 16KB weight chunks
  int tid  = threadIdx.x;
  int wv   = tid >> 5;
  int lane = tid & 31;
  int n0   = wv * 16;                // this wave's n-tile
  int half = lane >> 4;
  int nc   = n0 + (lane & 15);       // B-matrix column (N)
  const unsigned lb0 = lds_addr(&wlds[0][0]);
  const unsigned lb1 = lds_addr(&wlds[1][0]);

  int bI[CONV_MT], oyI[CONV_MT], oxI[CONV_MT];
#pragma unroll
  for (int i = 0; i < CONV_MT; ++i) {
    int mr  = (blockIdx.x * CONV_MT + i) * 16 + (lane & 15);
    bI[i]   = mr / (HN * WN);
    int rem = mr % (HN * WN);
    oyI[i]  = rem / WN;
    oxI[i]  = rem % WN;
  }

  // prologue: issue chunk 0 into buffer 0
#pragma unroll
  for (int i = 0; i < 4; ++i) {
    unsigned slot = (unsigned)tid + (unsigned)i * 256u;
    async_g2l_b128(lb0 + slot * 16u, slot * 16u, cw);
  }

  v8f acc[CONV_MT] = {};
  for (int kk = 0; kk < 18; ++kk) {  // 18 K-chunks of 32
    unsigned lcur = (kk & 1) ? lb1 : lb0;
    if (kk < 17) {                   // prefetch next chunk into other buffer
      unsigned lnext = (kk & 1) ? lb0 : lb1;
#pragma unroll
      for (int i = 0; i < 4; ++i) {
        unsigned slot = (unsigned)tid + (unsigned)i * 256u;
        async_g2l_b128(lnext + slot * 16u,
                       (unsigned)(kk + 1) * 16384u + slot * 16u, cw);
      }
      wait_async4();                 // in-order: chunk kk has landed
    } else {
      wait_async0();
    }
    __syncthreads();                 // all waves' async stores visible

    const float* wbuf = (kk & 1) ? &wlds[1][0] : &wlds[0][0];
    int tap = kk >> 1;
    int cinbase = (kk & 1) * 32;
    int ty = tap / 3, tx = tap % 3;

    Frag16 bf;                       // B fragment: shared by all m-tiles
#pragma unroll
    for (int v = 0; v < 8; ++v) {
      int kl = (v < 4) ? (half * 8 + 2 * v) : (16 + half * 8 + 2 * (v - 4));
      bf.u[2 * v]     = f2bf(wbuf[kl * COUT + nc]);
      bf.u[2 * v + 1] = f2bf(wbuf[(kl + 1) * COUT + nc]);
    }

#pragma unroll
    for (int i = 0; i < CONV_MT; ++i) {
      int iy = oyI[i] * 2 - 1 + ty;
      int ix = oxI[i] * 2 - 1 + tx;
      bool inb = (iy >= 0) && (iy < HH) && (ix >= 0) && (ix < WW);
      const float* src = fmap + (((long)bI[i] * HH + iy) * WW + ix) * CIN;
      Frag16 a;
#pragma unroll
      for (int v = 0; v < 8; ++v) {
        int kl = (v < 4) ? (half * 8 + 2 * v) : (16 + half * 8 + 2 * (v - 4));
        float f0 = 0.f, f1 = 0.f;
        if (inb) {                   // contiguous cin pair -> one b64 load
          float2 t2 = *(const float2*)(src + cinbase + kl);
          f0 = t2.x; f1 = t2.y;
        }
        a.u[2 * v]     = f2bf(f0);
        a.u[2 * v + 1] = f2bf(f1);
      }
      acc[i] = __builtin_amdgcn_wmma_f32_16x16x32_bf16(
          false, a.v, false, bf.v, (short)0, acc[i], false, false);
    }
    __syncthreads();                 // chunk consumed before its buffer reuse
  }

  int col = n0 + (lane & 15);
  float bias = cb[col];
#pragma unroll
  for (int i = 0; i < CONV_MT; ++i) {
#pragma unroll
    for (int r = 0; r < 8; ++r) {
      int row = (blockIdx.x * CONV_MT + i) * 16 + half * 8 + r;
      y[(long)row * COUT + col] = acc[i][r] + bias;
    }
  }
}

// ================= kernel 4: skip GEMM tok = x @ skip_w (WMMA bf16) =====
// skip_w (64x128 f32 = 32KB) staged once per block via async LDS loads and
// reused across SKIP_MT m-tiles per wave (block covers 128 rows).
__global__ void k_skip_wmma(const float* __restrict__ x,   // [MROWS,64]
                            const float* __restrict__ sw,  // [64,128]
                            float* __restrict__ tok) {     // [MROWS,128]
  __shared__ float swlds[CIN * COUT];              // 32KB
  int tid  = threadIdx.x;
  int wv   = tid >> 5;
  int lane = tid & 31;
  int n0   = wv * 16;
  int half = lane >> 4;
  int nc   = n0 + (lane & 15);
  const unsigned lbase = lds_addr(swlds);

  // ---- stage skip_w: 2048 x 16B slots, 8 per thread ----
#pragma unroll
  for (int i = 0; i < 8; ++i) {
    unsigned slot = (unsigned)tid + (unsigned)i * 256u;
    async_g2l_b128(lbase + slot * 16u, slot * 16u, sw);
  }
  wait_async0();
  __syncthreads();

  v8f acc[SKIP_MT] = {};
#pragma unroll
  for (int kc = 0; kc < CIN; kc += 32) {
    Frag16 bf;                       // B fragment shared across m-tiles
#pragma unroll
    for (int v = 0; v < 8; ++v) {
      int kl = (v < 4) ? (half * 8 + 2 * v) : (16 + half * 8 + 2 * (v - 4));
      int k = kc + kl;
      bf.u[2 * v]     = f2bf(swlds[k * COUT + nc]);
      bf.u[2 * v + 1] = f2bf(swlds[(k + 1) * COUT + nc]);
    }
#pragma unroll
    for (int i = 0; i < SKIP_MT; ++i) {
      long mr = (long)(blockIdx.x * SKIP_MT + i) * 16 + (lane & 15);
      const float* xrow = x + mr * CIN;
      Frag16 a;
#pragma unroll
      for (int v = 0; v < 8; ++v) {
        int kl = (v < 4) ? (half * 8 + 2 * v) : (16 + half * 8 + 2 * (v - 4));
        float2 t2 = *(const float2*)(xrow + kc + kl);
        a.u[2 * v]     = f2bf(t2.x);
        a.u[2 * v + 1] = f2bf(t2.y);
      }
      acc[i] = __builtin_amdgcn_wmma_f32_16x16x32_bf16(
          false, a.v, false, bf.v, (short)0, acc[i], false, false);
    }
  }
  int col = n0 + (lane & 15);
#pragma unroll
  for (int i = 0; i < SKIP_MT; ++i) {
#pragma unroll
    for (int r = 0; r < 8; ++r) {
      long row = (long)(blockIdx.x * SKIP_MT + i) * 16 + half * 8 + r;
      tok[row * COUT + col] = acc[i][r];
    }
  }
}

// ================= kernel 5: map2token denominator scatter =================
__global__ void k_m2t_allw(const float* __restrict__ loc,
                           const int* __restrict__ idx_agg,
                           const float* __restrict__ aggw,
                           float* __restrict__ allw) {
  int t = blockIdx.x * blockDim.x + threadIdx.x;
  if (t >= NPTS) return;
  int b = t / N0;
  int i4[4]; float w4[4];
  bilin4(loc[2 * t], loc[2 * t + 1], HN, WN, i4, w4);
  float aw = aggw[t];
  float vs = (w4[0] + w4[1] + w4[2] + w4[3]) * aw;
  atomicAdd(&allw[(long)b * N0 + idx_agg[t]], vs);
}

// ================= kernel 6: map2token gather + scatter into tok ===========
__global__ void k_m2t_scatter(const float* __restrict__ y,     // [B*NS,128]
                              const float* __restrict__ loc,
                              const int* __restrict__ idx_agg,
                              const float* __restrict__ aggw,
                              const float* __restrict__ allw,
                              float* __restrict__ tok) {
  int t = blockIdx.x;            // orig point
  int c = threadIdx.x;           // channel 0..127
  int b = t / N0;
  int i4[4]; float w4[4];
  bilin4(loc[2 * t], loc[2 * t + 1], HN, WN, i4, w4);
  int ia = idx_agg[t];
  float aw = aggw[t];
  float denom = allw[(long)b * N0 + ia] + EPS;
  float val = 0.f;
#pragma unroll
  for (int j = 0; j < 4; ++j) {
    float vn = w4[j] * aw / denom;
    val += vn * y[((long)b * NS + i4[j]) * COUT + c];
  }
  atomicAdd(&tok[((long)b * N0 + ia) * COUT + c], val);
}

// ================= kernel 7: BN reduction (sum / sumsq per channel) ========
__global__ void k_bn_reduce(const float* __restrict__ tok,
                            float* __restrict__ bsum,
                            float* __restrict__ bsq) {
  int c = threadIdx.x;                    // 128 threads, one channel each
  float s = 0.f, q = 0.f;
  for (long r = blockIdx.x; r < MROWS; r += gridDim.x) {
    float v = tok[r * COUT + c];
    s += v; q += v * v;
  }
  atomicAdd(&bsum[c], s);
  atomicAdd(&bsq[c], q);
}

// ================= kernel 8: BN apply + ReLU(tok) out + conf head =========
__global__ void k_bn_apply(float* __restrict__ tok,
                           const float* __restrict__ bsum,
                           const float* __restrict__ bsq,
                           const float* __restrict__ gamma,
                           const float* __restrict__ beta,
                           const float* __restrict__ conf_w,
                           const float* __restrict__ conf_b,
                           float* __restrict__ o_tok,
                           float* __restrict__ o_conf) {
  __shared__ float red[COUT];
  long row = blockIdx.x;
  int c = threadIdx.x;
  const float invM = 1.0f / (float)MROWS;
  float mu  = bsum[c] * invM;
  float var = bsq[c] * invM - mu * mu;
  float tn = (tok[row * COUT + c] - mu) * rsqrtf(var + BN_EPS) * gamma[c] + beta[c];
  tok[row * COUT + c]   = tn;                 // keep pre-relu for clustering
  o_tok[row * COUT + c] = fmaxf(tn, 0.f);     // relu(tok) output
  red[c] = tn * conf_w[c];
  __syncthreads();
  for (int s = COUT / 2; s > 0; s >>= 1) {
    if (c < s) red[c] += red[c + s];
    __syncthreads();
  }
  if (c == 0) o_conf[row] = red[0] + conf_b[0];
}

// ================= kernel 9: cluster denominator scatter + idx output ======
__global__ void k_clus_allw(const float* __restrict__ loc,
                            const int* __restrict__ idx_agg,
                            const float* __restrict__ conf,
                            float* __restrict__ allwc,
                            float* __restrict__ o_idx) {
  int t = blockIdx.x * blockDim.x + threadIdx.x;
  if (t >= NPTS) return;
  int b = t / N0;
  int idn = grid_idx(loc[2 * t], loc[2 * t + 1], HN, WN);
  o_idx[t] = (float)idn;                     // idx_agg_down (int as float)
  int ia = idx_agg[t];
  float tw = expf(conf[(long)b * N0 + ia]);
  atomicAdd(&allwc[(long)b * NS + idn], tw);
}

// ================= kernel 10: weighted cluster scatter into x_down ========
__global__ void k_clus_scatter(const float* __restrict__ tok,
                               const float* __restrict__ loc,
                               const int* __restrict__ idx_agg,
                               const float* __restrict__ conf,
                               const float* __restrict__ aggw,
                               const float* __restrict__ allwc,
                               float* __restrict__ o_xdown,
                               float* __restrict__ o_aw,
                               unsigned int* __restrict__ bmax) {
  int t = blockIdx.x;            // orig point
  int c = threadIdx.x;           // channel 0..127
  int b = t / N0;
  int idn = grid_idx(loc[2 * t], loc[2 * t + 1], HN, WN);
  int ia = idx_agg[t];
  float tw = expf(conf[(long)b * N0 + ia]);
  float nw = tw / (allwc[(long)b * NS + idn] + EPS);
  float v = tok[((long)b * N0 + ia) * COUT + c] * nw;
  atomicAdd(&o_xdown[((long)b * NS + idn) * COUT + c], v);
  if (c == 0) {
    float aw = aggw[t] * nw;                 // agg_weight * weight_t
    o_aw[t] = aw;
    atomicMax(&bmax[b], __float_as_uint(aw)); // aw >= 0 -> bits monotone
  }
}

// ================= kernel 11: finalize (relu x_down, normalize aw) ========
__global__ void k_finalize(float* __restrict__ o_xdown,
                           float* __restrict__ o_aw,
                           const unsigned int* __restrict__ bmax) {
  long t = (long)blockIdx.x * blockDim.x + threadIdx.x;
  const long nx = (long)BB * NS * COUT;
  if (t < nx) o_xdown[t] = fmaxf(o_xdown[t], 0.f);
  if (t < NPTS) {
    int b = (int)(t / N0);
    float m = __uint_as_float(bmax[b]);
    o_aw[t] = o_aw[t] / m;
  }
}

// ======================= launcher =======================
extern "C" void kernel_launch(void* const* d_in, const int* in_sizes, int n_in,
                              void* d_out, int out_size, void* d_ws, size_t ws_size,
                              hipStream_t stream) {
  const float* x       = (const float*)d_in[0];   // [B,N0,64]
  const float* loc     = (const float*)d_in[1];   // [B,N0,2]
  const int*   idx_agg = (const int*)  d_in[2];   // [B,N0]
  const float* aggw    = (const float*)d_in[3];   // [B,N0,1]
  const float* conv_w  = (const float*)d_in[4];   // [3,3,64,128]
  const float* conv_b  = (const float*)d_in[5];   // [128]
  const float* skip_w  = (const float*)d_in[6];   // [64,128]
  const float* gamma   = (const float*)d_in[7];   // [128]
  const float* beta    = (const float*)d_in[8];   // [128]
  const float* conf_w  = (const float*)d_in[9];   // [128,1]
  const float* conf_b  = (const float*)d_in[10];  // [1]

  // ---- workspace layout (floats) ----
  float* ws    = (float*)d_ws;
  float* cnt   = ws;                                   // B*H*W
  float* fmap  = cnt   + (long)BB * HH * WW;           // B*H*W*64
  float* y     = fmap  + (long)BB * HH * WW * CIN;     // B*NS*128
  float* allw  = y     + (long)BB * NS * COUT;         // B*N0
  float* tok   = allw  + (long)BB * N0;                // B*N0*128
  float* allwc = tok   + (long)BB * N0 * COUT;         // B*NS
  float* bsum  = allwc + (long)BB * NS;                // 128
  float* bsq   = bsum  + COUT;                         // 128
  unsigned int* bmax = (unsigned int*)(bsq + COUT);    // 32
  size_t ws_used = ((char*)(bmax + BB)) - (char*)d_ws;

  // ---- output layout ----
  float* out     = (float*)d_out;
  float* o_xdown = out;                                        // B*NS*128
  float* o_idx   = o_xdown + (long)BB * NS * COUT;             // B*N0
  float* o_aw    = o_idx   + (long)BB * N0;                    // B*N0
  float* o_tok   = o_aw    + (long)BB * N0;                    // B*N0*128
  float* o_conf  = o_tok   + (long)BB * N0 * COUT;             // B*N0

  // 0) zero scratch + outputs (graph-capturable)
  hipMemsetAsync(d_ws, 0, ws_used < ws_size ? ws_used : ws_size, stream);
  hipMemsetAsync(d_out, 0, (size_t)out_size * sizeof(float), stream);

  // 1) token2map: counts then weighted scatter
  k_count<<<(NPTS + 255) / 256, 256, 0, stream>>>(loc, cnt);
  k_scatter_fmap<<<NPTS, CIN, 0, stream>>>(x, loc, idx_agg, cnt, fmap);

  // 2) conv 3x3 s2 via WMMA implicit GEMM (pipelined async weight staging)
  k_conv_wmma<<<MCONV / 16 / CONV_MT, 256, 0, stream>>>(fmap, conv_w, conv_b, y);

  // 3) skip GEMM via WMMA (one-shot async weight staging)
  k_skip_wmma<<<MROWS / 16 / SKIP_MT, 256, 0, stream>>>(x, skip_w, tok);

  // 4) map2token: denominator then gather+scatter-add into tok
  k_m2t_allw<<<(NPTS + 255) / 256, 256, 0, stream>>>(loc, idx_agg, aggw, allw);
  k_m2t_scatter<<<NPTS, COUT, 0, stream>>>(y, loc, idx_agg, aggw, allw, tok);

  // 5) batchnorm (training stats) + relu(tok) + conf head
  k_bn_reduce<<<1024, COUT, 0, stream>>>(tok, bsum, bsq);
  k_bn_apply<<<MROWS, COUT, 0, stream>>>(tok, bsum, bsq, gamma, beta,
                                         conf_w, conf_b, o_tok, o_conf);

  // 6) confidence-weighted grid clustering
  k_clus_allw<<<(NPTS + 255) / 256, 256, 0, stream>>>(loc, idx_agg, o_conf, allwc, o_idx);
  k_clus_scatter<<<NPTS, COUT, 0, stream>>>(tok, loc, idx_agg, o_conf, aggw,
                                            allwc, o_xdown, o_aw, bmax);

  // 7) finalize: relu(x_down), aw_down /= per-batch max
  long nfin = (long)BB * NS * COUT;
  k_finalize<<<(unsigned)((nfin + 255) / 256), 256, 0, stream>>>(o_xdown, o_aw, bmax);
}
